// SPRGraphNet_88648124990373
// MI455X (gfx1250) — matrix operand order repaired and checked by reading, not compiled, not run on hardware
//
#include <hip/hip_runtime.h>
#include <hip/hip_bf16.h>

#define NNODES 100000   // multiple of 16 -> 6250 exact WMMA row tiles
#define NEDGES 1200000
#define NGRAPHS 2048
#define TB 256
#define WPACK_HALVES 4096   // 4 ntiles * 2 ksteps * 32 lanes * 16 halves

typedef __attribute__((ext_vector_type(16))) _Float16 v16h;
typedef __attribute__((ext_vector_type(8)))  float    v8f;

// ---------------- utility ----------------
__global__ void zero_f32_kernel(float* __restrict__ p, long long n) {
    long long t = (long long)blockIdx.x * blockDim.x + threadIdx.x;
    if (t < n) p[t] = 0.0f;
}

// ---------------- pack weight matrix into WMMA B-fragment layout ----------------
// out[((nt*2+ks)*32 + lane)*16 + e] = f16( W[k][ncol] ), k = ks*32 + (lane>>4)*8 + (e&7) + (e>>3)*16,
// ncol = nt*16 + (lane&15); rows k >= Klim are zero-padded (layer 1 has Klim=48).
__global__ void pack_w_kernel(const float* __restrict__ W, int Klim,
                              _Float16* __restrict__ out) {
    int t = blockIdx.x * blockDim.x + threadIdx.x;
    if (t >= WPACK_HALVES) return;
    int e    = t & 15;
    int lane = (t >> 4) & 31;
    int ks   = (t >> 9) & 1;
    int nt   = t >> 10;
    int k    = ks * 32 + (lane >> 4) * 8 + (e & 7) + (e >> 3) * 16;
    int ncol = nt * 16 + (lane & 15);
    out[t] = (k < Klim) ? (_Float16)W[k * 64 + ncol] : (_Float16)0.0f;
}

// ---------------- node embedding: h[n][0..63] (cols 48..63 zero-padded) ----------------
__global__ void embed_kernel(const int* __restrict__ x,
                             const float* __restrict__ shape_emb,
                             const float* __restrict__ color_emb,
                             const float* __restrict__ pos_emb,
                             float* __restrict__ h) {
    int t = blockIdx.x * blockDim.x + threadIdx.x;
    if (t >= NNODES * 64) return;
    int n = t >> 6, c = t & 63;
    float v = 0.0f;
    if (c < 16) {
        v = shape_emb[x[n * 3 + 0] * 16 + c];
    } else if (c < 32) {
        v = color_emb[x[n * 3 + 1] * 16 + (c - 16)];
    } else if (c < 48) {
        int p = x[n * 3 + 2];
        p = p < 0 ? 0 : (p > 19 ? 19 : p);
        v = pos_emb[p * 16 + (c - 32)];
    }
    h[t] = v;
}

// ---------------- degree: deg[dst[e]] += 1 ----------------
__global__ void deg_kernel(const int* __restrict__ dst, float* __restrict__ deg) {
    int e = blockIdx.x * blockDim.x + threadIdx.x;
    if (e < NEDGES) atomicAdd(&deg[dst[e]], 1.0f);
}

// ---------------- edge scatter: agg[dst][c] += h[src][c] ----------------
__global__ void scatter_kernel(const int* __restrict__ src, const int* __restrict__ dst,
                               const float* __restrict__ h, float* __restrict__ agg) {
    long long t = (long long)blockIdx.x * blockDim.x + threadIdx.x;
    if (t >= (long long)NEDGES * 64) return;
    int e = (int)(t >> 6), c = (int)(t & 63);
    atomicAdd(&agg[(long long)dst[e] * 64 + c], h[(long long)src[e] * 64 + c]);
}

// ---------------- mean: agg[n][c] /= max(deg[n],1) (in place) ----------------
__global__ void mean_kernel(float* __restrict__ agg, const float* __restrict__ deg) {
    int t = blockIdx.x * blockDim.x + threadIdx.x;
    if (t >= NNODES * 64) return;
    float d = deg[t >> 6];
    agg[t] = agg[t] / (d > 1.0f ? d : 1.0f);
}

// ---------------- fused SAGE layer GEMM via WMMA ----------------
// out = relu( Am @ Wl + As @ Wr + bias ); Am/As: NNODES x 64 f32 (padded);
// BL/BR: pre-packed f16 B fragments. One wave computes a 16x64 row strip.
// NNODES % 16 == 0 -> no tail handling, fully unconditional loads/stores.
__global__ __launch_bounds__(256) void sage_gemm_wmma(
    const float* __restrict__ Am, const float* __restrict__ As,
    const v16h* __restrict__ BL, const v16h* __restrict__ BR,
    const float* __restrict__ bias, float* __restrict__ out) {
    int lane = threadIdx.x & 31;
    int wave = threadIdx.x >> 5;
    int tile = blockIdx.x * 8 + wave;
    int rowBase = tile * 16;
    if (rowBase >= NNODES) return;  // uniform per-wave exit (tail waves of last block)

    int lg = lane >> 4;              // lane group 0/1
    int r = rowBase + (lane & 15);   // A row for this lane
    const float* rowM = Am + (size_t)r * 64;
    const float* rowS = As + (size_t)r * 64;

    // Preload A fragments: 2 sources x 2 k-steps.
    // 16-bit A 16x32 layout: lane l holds row l%16; halves 0..7 = K=lg*8+ks*32+0..7,
    // halves 8..15 = K=lg*8+ks*32+16..23 -> contiguous float4 pairs.
    v16h aM[2], aS[2];
#pragma unroll
    for (int ks = 0; ks < 2; ++ks) {
        int kb = ks * 32 + lg * 8;
        const float4* m0 = reinterpret_cast<const float4*>(rowM + kb);
        const float4* m1 = reinterpret_cast<const float4*>(rowM + kb + 16);
        const float4* s0 = reinterpret_cast<const float4*>(rowS + kb);
        const float4* s1 = reinterpret_cast<const float4*>(rowS + kb + 16);
        float4 mA = m0[0], mB = m0[1], mC = m1[0], mD = m1[1];
        float4 sA = s0[0], sB = s0[1], sC = s1[0], sD = s1[1];
        aM[ks][0]  = (_Float16)mA.x; aM[ks][1]  = (_Float16)mA.y;
        aM[ks][2]  = (_Float16)mA.z; aM[ks][3]  = (_Float16)mA.w;
        aM[ks][4]  = (_Float16)mB.x; aM[ks][5]  = (_Float16)mB.y;
        aM[ks][6]  = (_Float16)mB.z; aM[ks][7]  = (_Float16)mB.w;
        aM[ks][8]  = (_Float16)mC.x; aM[ks][9]  = (_Float16)mC.y;
        aM[ks][10] = (_Float16)mC.z; aM[ks][11] = (_Float16)mC.w;
        aM[ks][12] = (_Float16)mD.x; aM[ks][13] = (_Float16)mD.y;
        aM[ks][14] = (_Float16)mD.z; aM[ks][15] = (_Float16)mD.w;
        aS[ks][0]  = (_Float16)sA.x; aS[ks][1]  = (_Float16)sA.y;
        aS[ks][2]  = (_Float16)sA.z; aS[ks][3]  = (_Float16)sA.w;
        aS[ks][4]  = (_Float16)sB.x; aS[ks][5]  = (_Float16)sB.y;
        aS[ks][6]  = (_Float16)sB.z; aS[ks][7]  = (_Float16)sB.w;
        aS[ks][8]  = (_Float16)sC.x; aS[ks][9]  = (_Float16)sC.y;
        aS[ks][10] = (_Float16)sC.z; aS[ks][11] = (_Float16)sC.w;
        aS[ks][12] = (_Float16)sD.x; aS[ks][13] = (_Float16)sD.y;
        aS[ks][14] = (_Float16)sD.z; aS[ks][15] = (_Float16)sD.w;
    }

#pragma unroll
    for (int nt = 0; nt < 4; ++nt) {
        int ncol = nt * 16 + (lane & 15);
        v8f acc = {};
#pragma unroll
        for (int ks = 0; ks < 2; ++ks) {
            v16h bL = BL[(nt * 2 + ks) * 32 + lane];  // one 32B vector load each
            v16h bR = BR[(nt * 2 + ks) * 32 + lane];
            acc = __builtin_amdgcn_wmma_f32_16x16x32_f16(false, aM[ks], false, bL,
                                                         (short)0, acc, false, false);
            acc = __builtin_amdgcn_wmma_f32_16x16x32_f16(false, aS[ks], false, bR,
                                                         (short)0, acc, false, false);
        }
        float bv = bias[ncol];
#pragma unroll
        for (int i = 0; i < 8; ++i) {
            int row = rowBase + lg * 8 + i;  // C/D layout: vgpr i -> M = i + lg*8
            float v = acc[i] + bv;
            out[(size_t)row * 64 + ncol] = v > 0.0f ? v : 0.0f;
        }
    }
}

// ---------------- graph pooling ----------------
__global__ void pool_kernel(const float* __restrict__ h, const int* __restrict__ batch,
                            float* __restrict__ gsum) {
    int t = blockIdx.x * blockDim.x + threadIdx.x;
    if (t >= NNODES * 64) return;
    int n = t >> 6, c = t & 63;
    atomicAdd(&gsum[(long long)batch[n] * 64 + c], h[t]);
}

__global__ void gcnt_kernel(const int* __restrict__ batch, float* __restrict__ gcnt) {
    int n = blockIdx.x * blockDim.x + threadIdx.x;
    if (n < NNODES) atomicAdd(&gcnt[batch[n]], 1.0f);
}

// ---------------- classifier: out[g][j] = mean_g @ Wc + bc ----------------
__global__ void cls_kernel(const float* __restrict__ gsum, const float* __restrict__ gcnt,
                           const float* __restrict__ Wc, const float* __restrict__ bc,
                           float* __restrict__ out) {
    int t = blockIdx.x * blockDim.x + threadIdx.x;
    if (t >= NGRAPHS * 10) return;
    int g = t / 10, j = t % 10;
    float d = gcnt[g];
    d = d > 1.0f ? d : 1.0f;
    float inv = 1.0f / d;
    float acc = bc[j];
#pragma unroll
    for (int c = 0; c < 64; ++c)
        acc += (gsum[g * 64 + c] * inv) * Wc[c * 10 + j];
    out[t] = acc;
}

extern "C" void kernel_launch(void* const* d_in, const int* in_sizes, int n_in,
                              void* d_out, int out_size, void* d_ws, size_t ws_size,
                              hipStream_t stream) {
    const int*   x         = (const int*)d_in[0];
    const int*   edge_src  = (const int*)d_in[1];            // edge_index[0]
    const int*   edge_dst  = ((const int*)d_in[1]) + NEDGES; // edge_index[1]
    const int*   batch     = (const int*)d_in[2];
    // d_in[3] = num_graphs scalar (compile-time constant here)
    const float* shape_emb = (const float*)d_in[4];
    const float* color_emb = (const float*)d_in[5];
    const float* pos_emb   = (const float*)d_in[6];
    const float* W1_l      = (const float*)d_in[7];   // 48 x 64
    const float* W1_r      = (const float*)d_in[8];   // 48 x 64
    const float* b1        = (const float*)d_in[9];
    const float* W2_l      = (const float*)d_in[10];  // 64 x 64
    const float* W2_r      = (const float*)d_in[11];
    const float* b2        = (const float*)d_in[12];
    const float* Wc        = (const float*)d_in[13];  // 64 x 10
    const float* bc        = (const float*)d_in[14];
    float* out = (float*)d_out;

    // workspace layout (packed weights first: keeps 32B alignment for v16h loads)
    _Float16* wp1l = (_Float16*)d_ws;                 // 4096 halves each
    _Float16* wp1r = wp1l + WPACK_HALVES;
    _Float16* wp2l = wp1r + WPACK_HALVES;
    _Float16* wp2r = wp2l + WPACK_HALVES;
    float* h    = (float*)(wp2r + WPACK_HALVES);      // NNODES*64  (h0, reused as h2)
    float* agg  = h    + (size_t)NNODES * 64;         // NNODES*64  (agg/mean, reused)
    float* h1   = agg  + (size_t)NNODES * 64;         // NNODES*64
    float* deg  = h1   + (size_t)NNODES * 64;         // NNODES
    float* gsum = deg  + (size_t)NNODES;              // NGRAPHS*64
    float* gcnt = gsum + (size_t)NGRAPHS * 64;        // NGRAPHS (contiguous after gsum)

    const long long nNodeF = (long long)NNODES * 64;
    const long long nEdgeF = (long long)NEDGES * 64;
    const int gNodeF = (int)((nNodeF + TB - 1) / TB);
    const int gEdgeF = (int)((nEdgeF + TB - 1) / TB);
    const int gEdge  = (NEDGES + TB - 1) / TB;
    const int gPack  = (WPACK_HALVES + TB - 1) / TB;
    const int tiles  = NNODES / 16;                   // exact
    const int gGemm  = (tiles + 7) / 8;               // 8 waves per 256-thread block

    // ---- pack weights into WMMA B-fragment layout ----
    pack_w_kernel<<<gPack, TB, 0, stream>>>(W1_l, 48, wp1l);
    pack_w_kernel<<<gPack, TB, 0, stream>>>(W1_r, 48, wp1r);
    pack_w_kernel<<<gPack, TB, 0, stream>>>(W2_l, 64, wp2l);
    pack_w_kernel<<<gPack, TB, 0, stream>>>(W2_r, 64, wp2r);

    // ---- init accumulators ----
    zero_f32_kernel<<<gNodeF, TB, 0, stream>>>(agg, nNodeF);
    zero_f32_kernel<<<(NNODES + TB - 1) / TB, TB, 0, stream>>>(deg, NNODES);

    // ---- embedding ----
    embed_kernel<<<gNodeF, TB, 0, stream>>>(x, shape_emb, color_emb, pos_emb, h);

    // ---- degrees (shared by both layers) ----
    deg_kernel<<<gEdge, TB, 0, stream>>>(edge_dst, deg);

    // ---- layer 1 ----
    scatter_kernel<<<gEdgeF, TB, 0, stream>>>(edge_src, edge_dst, h, agg);
    mean_kernel<<<gNodeF, TB, 0, stream>>>(agg, deg);
    sage_gemm_wmma<<<gGemm, TB, 0, stream>>>(agg, h, (const v16h*)wp1l,
                                             (const v16h*)wp1r, b1, h1);

    // ---- layer 2 ----
    zero_f32_kernel<<<gNodeF, TB, 0, stream>>>(agg, nNodeF);
    scatter_kernel<<<gEdgeF, TB, 0, stream>>>(edge_src, edge_dst, h1, agg);
    mean_kernel<<<gNodeF, TB, 0, stream>>>(agg, deg);
    sage_gemm_wmma<<<gGemm, TB, 0, stream>>>(agg, h1, (const v16h*)wp2l,
                                             (const v16h*)wp2r, b2, h);

    // ---- pooling + classifier ----
    zero_f32_kernel<<<((NGRAPHS * 65) + TB - 1) / TB, TB, 0, stream>>>(
        gsum, (long long)NGRAPHS * 65);  // clears gsum + gcnt (contiguous)
    pool_kernel<<<gNodeF, TB, 0, stream>>>(h, batch, gsum);
    gcnt_kernel<<<(NNODES + TB - 1) / TB, TB, 0, stream>>>(batch, gcnt);
    cls_kernel<<<(NGRAPHS * 10 + TB - 1) / TB, TB, 0, stream>>>(gsum, gcnt, Wc, bc, out);
}
